// alphaLSTMNetwork_51926154609452
// MI455X (gfx1250) — matrix operand order repaired and checked by reading, not compiled, not run on hardware
//
#include <hip/hip_runtime.h>

// ---------------------------------------------------------------------------
// MI455X (gfx1250) persistent-LSTM implementation.
// Single workgroup (8 wave32), all recurrent state in LDS, all matmuls via
// v_wmma_f32_16x16x32_f16 with f32 accumulate. Weights pre-packed to f16 in
// WMMA B-fragment order in d_ws by a grid-parallel prep kernel.
// ---------------------------------------------------------------------------

typedef __attribute__((ext_vector_type(16))) _Float16 v16h;
typedef __attribute__((ext_vector_type(8)))  float    v8f;

#define NTHREADS 256
#define NWAVES   8

// problem dims
#define T_STEPS 4096
#define G_DIM   128
#define D_DIM   64
#define E_DIM   8
#define IN_DIM  192      // G + D
#define PRE_H   256
#define CI_DIM  128
#define CH_DIM  128
#define OUT_H   128
#define NF_DIM  64
#define OIN_DIM 448      // 2*CH + IN

// ---- LDS layout (floats then halfs) ---------------------------------------
// float region (element offsets)
#define F_X    0                 // 16*192  = 3072
#define F_H0   (F_X   + 3072)    // 16*256  = 4096
#define F_C0   (F_H0  + 4096)
#define F_H1   (F_C0  + 4096)
#define F_C1   (F_H1  + 4096)
#define F_PRE  (F_C1  + 4096)    // 16*128  = 2048
#define F_CH   (F_PRE + 2048)    // 2*128   = 256
#define F_CC   (F_CH  + 256)     // 2*128   = 256
#define F_OH   (F_CC  + 256)     // 16*128  = 2048
#define F_OC   (F_OH  + 2048)    // 16*128  = 2048
#define F_G    (F_OC  + 2048)    // 16*1024 = 16384
#define F_TOTAL (F_G + 16384)    // 42496 floats = 169984 B (32B aligned)
// half region (element offsets, after float region)
#define H_XP    0                // 16*192 = 3072
#define H_H0P   3072             // 16*256 = 4096
#define H_H1P   (H_H0P + 4096)
#define H_OHP   (H_H1P + 4096)   // 16*128 = 2048
#define H_OINP  (H_OHP + 2048)   // 16*448 = 7168
#define H_AX    (H_OINP + 7168)  // 16*128 = 2048
#define H_AH    (H_AX + 2048)    // 16*128 = 2048
#define H_TOTAL (H_AH + 2048)    // 24576 halfs = 49152 B
#define SMEM_BYTES (F_TOTAL*4 + H_TOTAL*2)   // 219136 B < 320 KB WGP LDS

struct Params {
  const float *features, *pre_h0, *pre_c0, *out_h0, *out_c0;
  const float *b_p0, *b_p1, *bfc, *b_c0, *b_c1, *b_o, *btar, *bdir;
  const _Float16 *w_ih_p0, *w_hh_p0, *w_ih_p1, *w_hh_p1, *w_fc;
  const _Float16 *w_ih_c0, *w_hh_c0, *w_ih_c1, *w_hh_c1;
  const _Float16 *w_ih_o, *w_hh_o, *w_tar, *w_dir;
  float *out_tar, *out_dir;
};

// ---------------------------------------------------------------------------
// Weight pack: W (Nsrc x K, row = output neuron) -> f16 tiles in WMMA
// B-fragment order. Tile (nt,kt) = 512 halfs; within tile: lane L holds
// column n = nt*16 + (L&15); K = kt*32 + ((L>=16)?16:0) + e (e = 0..15).
// ---------------------------------------------------------------------------
__global__ void pack_weights(const float* __restrict__ W, _Float16* __restrict__ out,
                             int Nsrc, int K, int total) {
  int idx = blockIdx.x * blockDim.x + threadIdx.x;
  if (idx >= total) return;
  int e    = idx & 15;
  int lane = (idx >> 4) & 31;
  int tile = idx >> 9;
  int kTiles = K >> 5;
  int nt = tile / kTiles;
  int kt = tile - nt * kTiles;
  int n = (nt << 4) + (lane & 15);
  int k = (kt << 5) + ((lane >> 4) << 4) + e;
  float v = (n < Nsrc) ? W[(size_t)n * K + k] : 0.0f;
  out[idx] = (_Float16)v;
}

// ---------------------------------------------------------------------------
// Device helpers
// ---------------------------------------------------------------------------

// Store one f32 activation element (m,k) into a packed f16 A-fragment buffer.
// A layout (contiguous, matches B packing): lanes 0-15: K 0-15,
// lanes 16-31: K 16-31.
__device__ __forceinline__ void st_pack(_Float16* ap, int m, int k, float v) {
  int kt = k >> 5, r = k & 31;
  int lane = (m & 15) | ((r >> 4) << 4);
  int e    = r & 15;
  ap[(kt << 9) + (lane << 4) + e] = (_Float16)v;
}

// Load one 512-half fragment tile: lane reads 16 contiguous halfs (32 B).
__device__ __forceinline__ v16h ld_tile(const _Float16* base, int tile) {
  int lane = threadIdx.x & 31;
  return *(const v16h*)(base + ((size_t)tile << 9) + (lane << 4));
}

// out(16 x Nout) = A1 * W1^T + A2 * W2^T + bias ; A packed in LDS, W packed in
// global (L2-resident). One 16x16 output tile per wave iteration.
__device__ __forceinline__ void gemm2(const _Float16* A1, int k1t, const _Float16* W1,
                                      const _Float16* A2, int k2t, const _Float16* W2,
                                      const float* bias, int biasLen,
                                      float* out, int Nout) {
  const int wave = threadIdx.x >> 5;
  const int lane = threadIdx.x & 31;
  const int nTiles = Nout >> 4;
  for (int nt = wave; nt < nTiles; nt += NWAVES) {
    v8f acc = {};
    for (int kt = 0; kt < k1t; ++kt) {
      v16h a = ld_tile(A1, kt);
      v16h b = ld_tile(W1, nt * k1t + kt);
      acc = __builtin_amdgcn_wmma_f32_16x16x32_f16(false, a, false, b,
                                                   (short)0, acc, false, false);
    }
    for (int kt = 0; kt < k2t; ++kt) {
      v16h a = ld_tile(A2, kt);
      v16h b = ld_tile(W2, nt * k2t + kt);
      acc = __builtin_amdgcn_wmma_f32_16x16x32_f16(false, a, false, b,
                                                   (short)0, acc, false, false);
    }
    // C/D layout: VGPR r -> row = r + (lane>=16 ? 8 : 0), col = lane&15
    const int row0 = (lane >> 4) << 3;
    const int col  = (nt << 4) + (lane & 15);
    const float bv = (col < biasLen) ? bias[col] : 0.0f;
#pragma unroll
    for (int r = 0; r < 8; ++r)
      out[(size_t)(row0 + r) * Nout + col] = acc[r] + bv;
  }
}

__device__ __forceinline__ float sigm(float x) { return 1.0f / (1.0f + expf(-x)); }

// Batched LSTM elementwise update: g is 16 x 4H (rows 0..7 valid).
__device__ __forceinline__ void lstm_update(const float* g, float* h, float* c,
                                            int H, _Float16* hp) {
  for (int i = threadIdx.x; i < E_DIM * H; i += NTHREADS) {
    int m = i / H, j = i - m * H;
    const float* gr = g + (size_t)m * 4 * H;
    float ii = sigm(gr[j]);
    float ff = sigm(gr[H + j]);
    float gg = tanhf(gr[2 * H + j]);
    float oo = sigm(gr[3 * H + j]);
    float c2 = ff * c[i] + ii * gg;
    float h2 = oo * tanhf(c2);
    c[i] = c2; h[i] = h2;
    st_pack(hp, m, j, h2);
  }
}

// Single-row comm-cell update with alpha blending (row 0 of g, 4*CH wide).
__device__ __forceinline__ void lstm_blend(const float* g, float* h, float* c,
                                           float alpha, _Float16* hp) {
  const int H = CH_DIM;
  for (int j = threadIdx.x; j < H; j += NTHREADS) {
    float ii = sigm(g[j]);
    float ff = sigm(g[H + j]);
    float gg = tanhf(g[2 * H + j]);
    float oo = sigm(g[3 * H + j]);
    float c2n = ff * c[j] + ii * gg;
    float h2n = oo * tanhf(c2n);
    float hb = (1.0f - alpha) * h[j] + alpha * h2n;
    float cb = (1.0f - alpha) * c[j] + alpha * c2n;
    h[j] = hb; c[j] = cb;
    if (hp) st_pack(hp, 0, j, hb);
  }
}

__device__ __forceinline__ void pack_vec(_Float16* dst, const float* src, int K) {
  for (int j = threadIdx.x; j < K; j += NTHREADS) st_pack(dst, 0, j, src[j]);
}

// ---------------------------------------------------------------------------
// Persistent kernel: one workgroup runs all 4096 timesteps.
// ---------------------------------------------------------------------------
__global__ void __launch_bounds__(NTHREADS, 1) lstm_persistent(Params P) {
  extern __shared__ char smem[];
  float*    fb = (float*)smem;
  _Float16* hb = (_Float16*)(smem + (size_t)F_TOTAL * 4);

  float* xv  = fb + F_X;   float* h0v = fb + F_H0;  float* c0v = fb + F_C0;
  float* h1v = fb + F_H1;  float* c1v = fb + F_C1;  float* prev = fb + F_PRE;
  float* chv = fb + F_CH;  float* ccv = fb + F_CC;
  float* ohv = fb + F_OH;  float* ocv = fb + F_OC;  float* gv  = fb + F_G;

  _Float16* xp   = hb + H_XP;   _Float16* h0p = hb + H_H0P;
  _Float16* h1p  = hb + H_H1P;  _Float16* ohp = hb + H_OHP;
  _Float16* oinp = hb + H_OINP; _Float16* aX  = hb + H_AX;
  _Float16* aH   = hb + H_AH;

  const int tid = threadIdx.x;

  // Zero all LDS (padded rows/garbage hygiene), then load initial states.
  for (int i = tid; i < SMEM_BYTES / 4; i += NTHREADS) ((unsigned int*)smem)[i] = 0u;
  __syncthreads();
  for (int i = tid; i < E_DIM * PRE_H; i += NTHREADS) {
    int m = i / PRE_H, k = i - m * PRE_H;
    float a0 = P.pre_h0[(size_t)(m * 2 + 0) * PRE_H + k];
    float a1 = P.pre_h0[(size_t)(m * 2 + 1) * PRE_H + k];
    h0v[i] = a0; h1v[i] = a1;
    st_pack(h0p, m, k, a0); st_pack(h1p, m, k, a1);
    c0v[i] = P.pre_c0[(size_t)(m * 2 + 0) * PRE_H + k];
    c1v[i] = P.pre_c0[(size_t)(m * 2 + 1) * PRE_H + k];
  }
  for (int i = tid; i < E_DIM * OUT_H; i += NTHREADS) {
    int m = i / OUT_H, k = i - m * OUT_H;
    float v = P.out_h0[i];
    ohv[i] = v; st_pack(ohp, m, k, v);
    ocv[i] = P.out_c0[i];
  }
  __syncthreads();

  for (int t = 0; t < T_STEPS; ++t) {
    // ---- stage 1: build x = [broadcast(group), ele] (8 x 192) -------------
    const float* feat = P.features + (size_t)t * (G_DIM + E_DIM * D_DIM);
    if (tid == 0 && t + 1 < T_STEPS)
      __builtin_prefetch(feat + (G_DIM + E_DIM * D_DIM), 0, 1);
    for (int i = tid; i < E_DIM * IN_DIM; i += NTHREADS) {
      int m = i / IN_DIM, k = i - m * IN_DIM;
      float v = (k < G_DIM) ? feat[k] : feat[G_DIM + m * D_DIM + (k - G_DIM)];
      xv[i] = v; st_pack(xp, m, k, v);
    }
    __syncthreads();

    // ---- pre LSTM layer 0: g = x@Wih^T + h0@Whh^T + b ---------------------
    gemm2(xp, IN_DIM / 32, P.w_ih_p0, h0p, PRE_H / 32, P.w_hh_p0,
          P.b_p0, 4 * PRE_H, gv, 4 * PRE_H);
    __syncthreads();
    lstm_update(gv, h0v, c0v, PRE_H, h0p);
    __syncthreads();

    // ---- pre LSTM layer 1 -------------------------------------------------
    gemm2(h0p, PRE_H / 32, P.w_ih_p1, h1p, PRE_H / 32, P.w_hh_p1,
          P.b_p1, 4 * PRE_H, gv, 4 * PRE_H);
    __syncthreads();
    lstm_update(gv, h1v, c1v, PRE_H, h1p);
    __syncthreads();

    // ---- FC + leaky ReLU --------------------------------------------------
    gemm2(h1p, PRE_H / 32, P.w_fc, (const _Float16*)nullptr, 0,
          (const _Float16*)nullptr, P.bfc, CI_DIM, gv, CI_DIM);
    __syncthreads();
    for (int i = tid; i < E_DIM * CI_DIM; i += NTHREADS) {
      float v = gv[i];
      prev[i] = v > 0.0f ? v : 0.05f * v;
    }
    for (int i = tid; i < 2 * CH_DIM; i += NTHREADS) { chv[i] = 0.0f; ccv[i] = 0.0f; }
    __syncthreads();

    // ---- comm rounds: 3 x scan over E rows x 2 dependent cells ------------
    float alpha = 1.0f;
    for (int rd = 0; rd < 3; ++rd) {
      for (int e2 = 0; e2 < E_DIM; ++e2) {
        pack_vec(aX, prev + (size_t)e2 * CI_DIM, CI_DIM);  // xin
        pack_vec(aH, chv, CH_DIM);                         // ch[0]
        __syncthreads();
        gemm2(aX, CI_DIM / 32, P.w_ih_c0, aH, CH_DIM / 32, P.w_hh_c0,
              P.b_c0, 4 * CH_DIM, gv, 4 * CH_DIM);
        __syncthreads();
        lstm_blend(gv, chv, ccv, alpha, aX);               // h0b -> aX
        pack_vec(aH, chv + CH_DIM, CH_DIM);                // ch[1]
        __syncthreads();
        gemm2(aX, CH_DIM / 32, P.w_ih_c1, aH, CH_DIM / 32, P.w_hh_c1,
              P.b_c1, 4 * CH_DIM, gv, 4 * CH_DIM);
        __syncthreads();
        lstm_blend(gv, chv + CH_DIM, ccv + CH_DIM, alpha, (_Float16*)nullptr);
        __syncthreads();
      }
      alpha *= 0.333f;
    }

    // ---- out cell input: [group_dec(=cc flat, 256), x(192)] ---------------
    for (int i = tid; i < E_DIM * OIN_DIM; i += NTHREADS) {
      int m = i / OIN_DIM, k = i - m * OIN_DIM;
      float v = (k < 2 * CH_DIM) ? ccv[k] : xv[m * IN_DIM + (k - 2 * CH_DIM)];
      st_pack(oinp, m, k, v);
    }
    __syncthreads();
    gemm2(oinp, OIN_DIM / 32, P.w_ih_o, ohp, OUT_H / 32, P.w_hh_o,
          P.b_o, 4 * OUT_H, gv, 4 * OUT_H);
    __syncthreads();
    lstm_update(gv, ohv, ocv, OUT_H, ohp);
    __syncthreads();

    // ---- heads ------------------------------------------------------------
    gemm2(ohp, OUT_H / 32, P.w_tar, (const _Float16*)nullptr, 0,
          (const _Float16*)nullptr, P.btar, NF_DIM, gv, NF_DIM);        // 16x64
    gemm2(ohp, OUT_H / 32, P.w_dir, (const _Float16*)nullptr, 0,
          (const _Float16*)nullptr, P.bdir, 3, gv + 1024, 16);          // 16x16
    __syncthreads();

    // ---- softmax + store --------------------------------------------------
    if (tid < E_DIM) {
      const float* row = gv + (size_t)tid * NF_DIM;
      float mx = row[0];
      for (int j = 1; j < NF_DIM; ++j) mx = fmaxf(mx, row[j]);
      float s = 0.0f;
      for (int j = 0; j < NF_DIM; ++j) s += expf(row[j] - mx);
      float inv = 1.0f / s;
      float* o = P.out_tar + ((size_t)t * E_DIM + tid) * NF_DIM;
      for (int j = 0; j < NF_DIM; ++j) o[j] = expf(row[j] - mx) * inv;
    } else if (tid < 2 * E_DIM) {
      int m = tid - E_DIM;
      const float* row = gv + 1024 + (size_t)m * 16;
      float mx = fmaxf(row[0], fmaxf(row[1], row[2]));
      float e0 = expf(row[0] - mx), e1 = expf(row[1] - mx), e2 = expf(row[2] - mx);
      float inv = 1.0f / (e0 + e1 + e2);
      float* o = P.out_dir + ((size_t)t * E_DIM + m) * 3;
      o[0] = e0 * inv; o[1] = e1 * inv; o[2] = e2 * inv;
    }
    __syncthreads();
  }
}

// ---------------------------------------------------------------------------
// Host launch
// ---------------------------------------------------------------------------
extern "C" void kernel_launch(void* const* d_in, const int* in_sizes, int n_in,
                              void* d_out, int out_size, void* d_ws, size_t ws_size,
                              hipStream_t stream) {
  (void)in_sizes; (void)n_in; (void)out_size; (void)ws_size;
  _Float16* ws = (_Float16*)d_ws;

  struct WD { int idx, Npad, Nsrc, K; };
  const WD wd[13] = {
    {5, 1024, 1024, 192},  // Wih_p0
    {6, 1024, 1024, 256},  // Whh_p0
    {8, 1024, 1024, 256},  // Wih_p1
    {9, 1024, 1024, 256},  // Whh_p1
    {11, 128, 128, 256},   // Wfc
    {13, 512, 512, 128},   // Wih_c0
    {14, 512, 512, 128},   // Whh_c0
    {16, 512, 512, 128},   // Wih_c1
    {17, 512, 512, 128},   // Whh_c1
    {19, 512, 512, 448},   // Wih_o
    {20, 512, 512, 128},   // Whh_o
    {22, 64, 64, 128},     // Wtar
    {24, 16, 3, 128},      // Wdir (rows padded 3->16 with zeros)
  };

  _Float16* wp[13];
  size_t off = 0;
  for (int i = 0; i < 13; ++i) {
    wp[i] = ws + off;
    int total = wd[i].Npad * wd[i].K;
    pack_weights<<<(total + 255) / 256, 256, 0, stream>>>(
        (const float*)d_in[wd[i].idx], wp[i], wd[i].Nsrc, wd[i].K, total);
    off += (size_t)total;
  }

  Params P;
  P.features = (const float*)d_in[0];
  P.pre_h0   = (const float*)d_in[1];
  P.pre_c0   = (const float*)d_in[2];
  P.out_h0   = (const float*)d_in[3];
  P.out_c0   = (const float*)d_in[4];
  P.b_p0 = (const float*)d_in[7];
  P.b_p1 = (const float*)d_in[10];
  P.bfc  = (const float*)d_in[12];
  P.b_c0 = (const float*)d_in[15];
  P.b_c1 = (const float*)d_in[18];
  P.b_o  = (const float*)d_in[21];
  P.btar = (const float*)d_in[23];
  P.bdir = (const float*)d_in[25];
  P.w_ih_p0 = wp[0];  P.w_hh_p0 = wp[1];
  P.w_ih_p1 = wp[2];  P.w_hh_p1 = wp[3];
  P.w_fc    = wp[4];
  P.w_ih_c0 = wp[5];  P.w_hh_c0 = wp[6];
  P.w_ih_c1 = wp[7];  P.w_hh_c1 = wp[8];
  P.w_ih_o  = wp[9];  P.w_hh_o  = wp[10];
  P.w_tar   = wp[11]; P.w_dir   = wp[12];
  P.out_tar = (float*)d_out;
  P.out_dir = (float*)d_out + (size_t)T_STEPS * E_DIM * NF_DIM;

  lstm_persistent<<<1, NTHREADS, SMEM_BYTES, stream>>>(P);
}
